// GIN_32684701123330
// MI455X (gfx1250) — compile-verified, matrix operand order checked by reading, not run on hardware
//
#include <hip/hip_runtime.h>
#include <hip/hip_bf16.h>

#define N_NODES 100000
#define N_EDGES 3200000
#define N_FEAT  128
#define DIM     256
#define N_CLS   16
#define BN_EPS  1e-5f

typedef __attribute__((ext_vector_type(16))) __bf16 v16bf;
typedef __attribute__((ext_vector_type(8)))  float  v8f;

// ---------------------------------------------------------------------------
// Edge scatter: agg[dst][:] += h[src][:]   (native f32 atomics, agent scope)
// One thread handles 4 consecutive columns of one edge.
// ---------------------------------------------------------------------------
__global__ void __launch_bounds__(256) scatter_add_kernel(
    const float* __restrict__ h, const int* __restrict__ ei,
    float* __restrict__ agg, int dim)
{
    size_t tid = (size_t)blockIdx.x * blockDim.x + threadIdx.x;
    int d4 = dim >> 2;
    size_t e = tid / (size_t)d4;
    if (e >= (size_t)N_EDGES) return;
    int q = (int)(tid - e * (size_t)d4);
    int s = ei[e];
    int d = ei[(size_t)N_EDGES + e];
    const float4 v = *(const float4*)(h + (size_t)s * dim + 4 * q);
    float* a = agg + (size_t)d * dim + 4 * q;
    __hip_atomic_fetch_add(a + 0, v.x, __ATOMIC_RELAXED, __HIP_MEMORY_SCOPE_AGENT);
    __hip_atomic_fetch_add(a + 1, v.y, __ATOMIC_RELAXED, __HIP_MEMORY_SCOPE_AGENT);
    __hip_atomic_fetch_add(a + 2, v.z, __ATOMIC_RELAXED, __HIP_MEMORY_SCOPE_AGENT);
    __hip_atomic_fetch_add(a + 3, v.w, __ATOMIC_RELAXED, __HIP_MEMORY_SCOPE_AGENT);
}

// ---------------------------------------------------------------------------
// (h + agg) -> bf16 A operand
// ---------------------------------------------------------------------------
__global__ void __launch_bounds__(256) prep_a16_kernel(
    const float* __restrict__ h, const float* __restrict__ agg,
    __bf16* __restrict__ out, size_t n)
{
    size_t i = (size_t)blockIdx.x * blockDim.x + threadIdx.x;
    if (i >= n) return;
    out[i] = (__bf16)(h[i] + agg[i]);
}

// f32 -> bf16 (for final FC input)
__global__ void __launch_bounds__(256) cvt_bf16_kernel(
    const float* __restrict__ h, __bf16* __restrict__ out, size_t n)
{
    size_t i = (size_t)blockIdx.x * blockDim.x + threadIdx.x;
    if (i >= n) return;
    out[i] = (__bf16)h[i];
}

// ---------------------------------------------------------------------------
// Pack f32 weight (K x N, row-major) into WMMA B-fragment layout:
// dst[ ((kt*NT + nt)*32 + lane)*16 + e ] where per-lane K index follows the
// CDNA5 16-bit operand table: K = kt*32 + (e&7) + 16*(e>>3) + 8*(lane>=16),
// N = nt*16 + (lane&15).
// ---------------------------------------------------------------------------
__global__ void __launch_bounds__(256) pack_w_kernel(
    const float* __restrict__ W, __bf16* __restrict__ out, int K, int N)
{
    int tid = blockIdx.x * blockDim.x + threadIdx.x;
    if (tid >= K * N) return;
    int e    = tid & 15;
    int lane = (tid >> 4) & 31;
    int rest = tid >> 9;
    int NT   = N >> 4;
    int nt   = rest % NT;
    int kt   = rest / NT;
    int kk = kt * 32 + (e & 7) + ((e >> 3) << 4) + ((lane >> 4) << 3);
    int nn = nt * 16 + (lane & 15);
    out[tid] = (__bf16)W[(size_t)kk * N + nn];
}

// ---------------------------------------------------------------------------
// WMMA GEMM: C[MxN] = A[MxK] @ B[KxN] (+bias, optional ReLU).
// One wave computes one 16x16 tile; K-loop of v_wmma_f32_16x16x32_bf16.
// A row-major bf16 (two aligned uint4 loads per fragment); B pre-packed.
// Writes f32 and/or bf16 outputs.
// ---------------------------------------------------------------------------
__global__ void __launch_bounds__(256) gemm_wmma_kernel(
    const __bf16* __restrict__ A, const __bf16* __restrict__ Bp,
    const float* __restrict__ bias,
    float* __restrict__ Cf, __bf16* __restrict__ Cb,
    int M, int N, int K, int relu)
{
    int wid = (int)(((size_t)blockIdx.x * blockDim.x + threadIdx.x) >> 5);
    int NT = N >> 4;
    int total = (M >> 4) * NT;
    if (wid >= total) return;           // whole wave exits together
    int nt = wid % NT;
    int mt = wid / NT;
    int lane = threadIdx.x & 31;
    int half = lane >> 4;
    int mr   = lane & 15;

    v8f acc = {};
    const __bf16* arow = A + (size_t)(mt * 16 + mr) * K + 8 * half;
    const __bf16* bp   = Bp + ((size_t)nt * 32 + lane) * 16;
    size_t bstep = (size_t)NT * 512;    // one kt step in packed B

    union { v16bf v; uint4 q[2]; } af, bfg;
    for (int k0 = 0; k0 < K; k0 += 32) {
        af.q[0]  = *(const uint4*)(arow + k0);
        af.q[1]  = *(const uint4*)(arow + k0 + 16);
        bfg.q[0] = *(const uint4*)(bp);
        bfg.q[1] = *(const uint4*)(bp + 8);
        bp += bstep;
        acc = __builtin_amdgcn_wmma_f32_16x16x32_bf16(
            false, af.v, false, bfg.v, (short)0, acc, false, false);
    }

    int n = nt * 16 + (lane & 15);
    float bv = bias ? bias[n] : 0.0f;
#pragma unroll
    for (int r = 0; r < 8; ++r) {
        float v = acc[r] + bv;
        if (relu) v = fmaxf(v, 0.0f);
        size_t row = (size_t)(mt * 16 + r + 8 * half);
        if (Cf) Cf[row * N + n] = v;
        if (Cb) Cb[row * N + n] = (__bf16)v;
    }
}

// ---------------------------------------------------------------------------
// BatchNorm (training-mode, biased variance): pass 1 = per-column sums
// ---------------------------------------------------------------------------
__global__ void __launch_bounds__(256) bn_stats_kernel(
    const float* __restrict__ t, float* __restrict__ stats, int rpb)
{
    int col = threadIdx.x;              // 256 columns
    int r0 = blockIdx.x * rpb;
    int r1 = r0 + rpb; if (r1 > N_NODES) r1 = N_NODES;
    float s = 0.f, s2 = 0.f;
    for (int r = r0; r < r1; ++r) {
        float v = t[(size_t)r * DIM + col];
        s += v; s2 += v * v;
    }
    __hip_atomic_fetch_add(stats + col,       s,  __ATOMIC_RELAXED, __HIP_MEMORY_SCOPE_AGENT);
    __hip_atomic_fetch_add(stats + DIM + col, s2, __ATOMIC_RELAXED, __HIP_MEMORY_SCOPE_AGENT);
}

__global__ void __launch_bounds__(256) bn_apply_kernel(
    const float* __restrict__ t, const float* __restrict__ stats,
    const float* __restrict__ g, const float* __restrict__ be,
    float* __restrict__ out, size_t n)
{
    size_t i = (size_t)blockIdx.x * blockDim.x + threadIdx.x;
    if (i >= n) return;
    int c = (int)(i & (DIM - 1));
    const float inv = 1.0f / (float)N_NODES;
    float m = stats[c] * inv;
    float v = stats[DIM + c] * inv - m * m;
    out[i] = g[c] * (t[i] - m) * rsqrtf(v + BN_EPS) + be[c];
}

// ---------------------------------------------------------------------------
// log_softmax over 16 classes per node
// ---------------------------------------------------------------------------
__global__ void __launch_bounds__(256) log_softmax_kernel(
    const float* __restrict__ logits, float* __restrict__ out)
{
    int i = blockIdx.x * blockDim.x + threadIdx.x;
    if (i >= N_NODES) return;
    const float* p = logits + (size_t)i * N_CLS;
    float mx = p[0];
#pragma unroll
    for (int c = 1; c < N_CLS; ++c) mx = fmaxf(mx, p[c]);
    float s = 0.f;
#pragma unroll
    for (int c = 0; c < N_CLS; ++c) s += __expf(p[c] - mx);
    float ls = __logf(s) + mx;
    float* o = out + (size_t)i * N_CLS;
#pragma unroll
    for (int c = 0; c < N_CLS; ++c) o[c] = p[c] - ls;
}

// ---------------------------------------------------------------------------
// Host orchestration
// ---------------------------------------------------------------------------
static inline unsigned div_up(size_t a, size_t b) { return (unsigned)((a + b - 1) / b); }

extern "C" void kernel_launch(void* const* d_in, const int* in_sizes, int n_in,
                              void* d_out, int out_size, void* d_ws, size_t ws_size,
                              hipStream_t stream)
{
    const float* x   = (const float*)d_in[0];
    const int*   ei  = (const int*)d_in[1];
    const float* w1a = (const float*)d_in[2];  const float* b1a = (const float*)d_in[3];
    const float* w1b = (const float*)d_in[4];  const float* b1b = (const float*)d_in[5];
    const float* g1  = (const float*)d_in[6];  const float* be1 = (const float*)d_in[7];
    const float* w2a = (const float*)d_in[8];  const float* b2a = (const float*)d_in[9];
    const float* w2b = (const float*)d_in[10]; const float* b2b = (const float*)d_in[11];
    const float* g2  = (const float*)d_in[12]; const float* be2 = (const float*)d_in[13];
    const float* w3a = (const float*)d_in[14]; const float* b3a = (const float*)d_in[15];
    const float* w3b = (const float*)d_in[16]; const float* b3b = (const float*)d_in[17];
    const float* wfc = (const float*)d_in[18]; const float* bfc = (const float*)d_in[19];

    // workspace carve (256B aligned)
    char* p = (char*)d_ws;
    auto carve = [&](size_t bytes) -> void* {
        void* r = (void*)p;
        p += (bytes + 255) & ~(size_t)255;
        return r;
    };
    float*  agg  = (float*) carve((size_t)N_NODES * DIM * sizeof(float));
    float*  tbuf = (float*) carve((size_t)N_NODES * DIM * sizeof(float));
    float*  hbuf = (float*) carve((size_t)N_NODES * DIM * sizeof(float));
    __bf16* a16  = (__bf16*)carve((size_t)N_NODES * DIM * sizeof(__bf16));
    __bf16* z16  = (__bf16*)carve((size_t)N_NODES * DIM * sizeof(__bf16));
    __bf16* wpA  = (__bf16*)carve((size_t)DIM * DIM * sizeof(__bf16));
    __bf16* wpB  = (__bf16*)carve((size_t)DIM * DIM * sizeof(__bf16));
    float*  stats= (float*) carve(2 * DIM * sizeof(float));

    const unsigned B = 256;

    auto run_gemm = [&](const __bf16* A, const __bf16* Bp, const float* bias,
                        float* Cf, __bf16* Cb, int M, int N, int K, int relu) {
        unsigned waves = (unsigned)((M >> 4) * (N >> 4));
        unsigned blocks = div_up((size_t)waves * 32, B);
        gemm_wmma_kernel<<<blocks, B, 0, stream>>>(A, Bp, bias, Cf, Cb, M, N, K, relu);
    };

    auto run_layer = [&](const float* hin, int dimIn,
                         const float* wa, const float* ba,
                         const float* wb, const float* bb,
                         const float* g, const float* be,
                         bool doBN, float* hout) {
        size_t nIn = (size_t)N_NODES * dimIn;
        hipMemsetAsync(agg, 0, nIn * sizeof(float), stream);
        scatter_add_kernel<<<div_up((size_t)N_EDGES * (dimIn >> 2), B), B, 0, stream>>>(
            hin, ei, agg, dimIn);
        prep_a16_kernel<<<div_up(nIn, B), B, 0, stream>>>(hin, agg, a16, nIn);
        pack_w_kernel<<<div_up((size_t)dimIn * DIM, B), B, 0, stream>>>(wa, wpA, dimIn, DIM);
        pack_w_kernel<<<div_up((size_t)DIM * DIM, B), B, 0, stream>>>(wb, wpB, DIM, DIM);
        // z = relu(A @ Wa + ba)  -> bf16
        run_gemm(a16, wpA, ba, nullptr, z16, N_NODES, DIM, dimIn, 1);
        // conv out = relu(z @ Wb + bb) -> f32 (to tbuf if BN, else straight to hout)
        run_gemm(z16, wpB, bb, doBN ? tbuf : hout, nullptr, N_NODES, DIM, DIM, 1);
        if (doBN) {
            hipMemsetAsync(stats, 0, 2 * DIM * sizeof(float), stream);
            const int grid = 512;
            const int rpb = (N_NODES + grid - 1) / grid;
            bn_stats_kernel<<<grid, DIM, 0, stream>>>(tbuf, stats, rpb);
            size_t n = (size_t)N_NODES * DIM;
            bn_apply_kernel<<<div_up(n, B), B, 0, stream>>>(tbuf, stats, g, be, hout, n);
        }
    };

    // layer 1 (input x, K=128), layer 2, layer 3 (no BN)
    run_layer(x,    N_FEAT, w1a, b1a, w1b, b1b, g1, be1, true,  hbuf);
    run_layer(hbuf, DIM,    w2a, b2a, w2b, b2b, g2, be2, true,  hbuf);
    run_layer(hbuf, DIM,    w3a, b3a, w3b, b3b, nullptr, nullptr, false, hbuf);

    // final classifier: logits = h @ wfc + bfc  (WMMA, N=16), then log_softmax
    size_t nH = (size_t)N_NODES * DIM;
    cvt_bf16_kernel<<<div_up(nH, B), B, 0, stream>>>(hbuf, a16, nH);
    pack_w_kernel<<<div_up((size_t)DIM * N_CLS, B), B, 0, stream>>>(wfc, wpA, DIM, N_CLS);
    run_gemm(a16, wpA, bfc, tbuf, nullptr, N_NODES, N_CLS, DIM, 0);
    log_softmax_kernel<<<div_up(N_NODES, B), B, 0, stream>>>(tbuf, (float*)d_out);
}
